// TemporalGCN_56324201120328
// MI455X (gfx1250) — matrix operand order repaired and checked by reading, not compiled, not run on hardware
//
#include <hip/hip_runtime.h>
#include <math.h>

// ---------------------------------------------------------------------------
// TemporalGCN + LSTM for MI455X (gfx1250, wave32, WMMA)
//
//   s   = A @ 1                                  (rowsum kernel)
//   Y   = A @ Xpad          (512 x 512 GEMM, bf16 WMMA, f32 accum)
//   Inp[(n,j),t] = relu(w_gc[j]*Y[n,t] + b_gc[j]*s[n])   (expand kernel, bf16)
//   G   = W_ih @ Inp        (2048 x 512 x K=16384 GEMM, bf16 WMMA) -- dominant
//   sequential LSTM over t using precomputed G  (persistent kernel, grid barrier)
// ---------------------------------------------------------------------------

typedef __attribute__((ext_vector_type(16))) __bf16       v16bf;
typedef __attribute__((ext_vector_type(8)))  float        v8f;
typedef __attribute__((ext_vector_type(8)))  unsigned int v8u;
typedef __attribute__((ext_vector_type(4)))  unsigned int v4u;

#define N_NODES 512
#define T_STEPS 500
#define T_PAD   512
#define HGDIM   32
#define RDIM    512
#define KBIG    16384   // N_NODES * HGDIM

__device__ __forceinline__ unsigned short f2bf(float f) {
  return __builtin_bit_cast(unsigned short, (__bf16)f);   // hw cvt where available
}

__device__ __forceinline__ float sigmoidf_(float x) { return 1.0f / (1.0f + expf(-x)); }

__device__ __forceinline__ float wave_reduce(float p) {
#pragma unroll
  for (int off = 16; off > 0; off >>= 1) p += __shfl_down(p, off, 32);
  return p;
}

// ---------------------------------------------------------------------------
// init: zero h buffer + grid-barrier state (ws is poisoned by harness)
// ---------------------------------------------------------------------------
__global__ void init_kernel(float* __restrict__ hbuf, int* __restrict__ bar) {
  int i = threadIdx.x;
  if (i < RDIM) hbuf[i] = 0.0f;
  if (i == 0) { bar[0] = 0; bar[1] = 0; }
}

// ---------------------------------------------------------------------------
// Xpad(512x512 bf16) = pad(x 512x500 f32), zero-padded columns
// ---------------------------------------------------------------------------
__global__ void convert_x_kernel(const float* __restrict__ x,
                                 unsigned short* __restrict__ Xbf) {
  int idx = blockIdx.x * 256 + threadIdx.x;   // 0 .. 512*512-1
  int k = idx >> 9;
  int t = idx & 511;
  Xbf[idx] = (t < T_STEPS) ? f2bf(x[(size_t)k * T_STEPS + t]) : (unsigned short)0;
}

// ---------------------------------------------------------------------------
// s[n] = sum_k A[n,k]    (A @ ones)
// ---------------------------------------------------------------------------
__global__ void rowsum_kernel(const float* __restrict__ A, float* __restrict__ s) {
  __shared__ float red[256];
  int n = blockIdx.x;
  float p = 0.0f;
  for (int k = threadIdx.x; k < N_NODES; k += 256) p += A[(size_t)n * N_NODES + k];
  red[threadIdx.x] = p;
  __syncthreads();
  for (int off = 128; off > 0; off >>= 1) {
    if (threadIdx.x < off) red[threadIdx.x] += red[threadIdx.x + off];
    __syncthreads();
  }
  if (threadIdx.x == 0) s[n] = red[0];
}

// ---------------------------------------------------------------------------
// Inp[(n*32+j), t] = relu(w_gc[j]*Y[n,t] + b_gc[j]*s[n])  -> bf16, 16384 x 512
// ---------------------------------------------------------------------------
__global__ void expand_kernel(const float* __restrict__ Y, const float* __restrict__ s,
                              const float* __restrict__ Wgc, const float* __restrict__ bgc,
                              unsigned short* __restrict__ Inp) {
  int t = blockIdx.x * blockDim.x + threadIdx.x;   // 0..511
  int n = blockIdx.y;                              // 0..511
  float y  = Y[(size_t)n * T_PAD + t];
  float sv = s[n];
#pragma unroll
  for (int j = 0; j < HGDIM; ++j) {
    float v = Wgc[j] * y + bgc[j] * sv;
    v = v > 0.0f ? v : 0.0f;
    Inp[((size_t)n * HGDIM + j) * T_PAD + t] = f2bf(v);
  }
}

// ---------------------------------------------------------------------------
// C(f32) = A(f32 -> bf16) x B(bf16) GEMM with v_wmma_f32_16x16x32_bf16.
// Block tile 128x64, BK=64 (2 K-steps, 8 WMMA per wave per staging round),
// 256 threads = 8 waves, each wave owns a 32x32 output (2x2 WMMA tiles).
// Global->register loads for tile kt+BK are issued before the WMMAs of tile
// kt so loads overlap compute (s_wait_loadcnt lands at next LDS store).
// ---------------------------------------------------------------------------
#define BM 128
#define BN 64
#define BK 64
#define LDA_S 72   // 64 + 8 pad shorts: 144B row stride (16B aligned, bank-spread)
#define LDB_S 72

__global__ __launch_bounds__(256)
void gemm_bf16_kernel(const float* __restrict__ Ag, const unsigned short* __restrict__ Bg,
                      float* __restrict__ Cg, int K, int lda, int ldb, int ldc) {
  __shared__ unsigned short As[BM * LDA_S];   // [m][k], k contiguous
  __shared__ unsigned short Bs[BN * LDB_S];   // transposed: [n][k], k contiguous

  const int tid  = threadIdx.x;
  const int wave = tid >> 5;
  const int lane = tid & 31;
  const int wm   = (wave & 3) * 32;   // wave row offset in block tile
  const int wn   = (wave >> 2) * 32;  // wave col offset in block tile
  const int bm   = blockIdx.x * BM;
  const int bn   = blockIdx.y * BN;
  const int half = lane >> 4;
  const int l15  = lane & 15;

  // staging coordinates (fixed per thread)
  const int ar = tid >> 1;            // A row for L2 prefetch: tid>>1 => 0..127, two threads per row
  const int br0 = tid >> 3;           // B row for i=0
  const int bc  = (tid & 7) * 8;      // B col group

  const v8f vzero = {0.f, 0.f, 0.f, 0.f, 0.f, 0.f, 0.f, 0.f};
  v8f acc[2][2];
  acc[0][0] = vzero; acc[0][1] = vzero; acc[1][0] = vzero; acc[1][1] = vzero;

  float4 areg[8];
  v4u    breg[2];

  // ---- prologue: load tile kt=0 into registers ----
#pragma unroll
  for (int i = 0; i < 8; ++i) {
    int fi = tid + 256 * i;
    int r  = fi >> 4;                 // 0..127
    int c  = (fi & 15) * 4;           // 0..60
    areg[i] = *(const float4*)(Ag + (size_t)(bm + r) * lda + c);
  }
#pragma unroll
  for (int i = 0; i < 2; ++i) {
    int fi = tid + 256 * i;
    int r  = fi >> 3;                 // 0..63
    int c  = (fi & 7) * 8;
    breg[i] = *(const v4u*)(Bg + (size_t)r * ldb + bn + c);
  }

  for (int kt = 0; kt < K; kt += BK) {
    __syncthreads();
    // ---- store staged registers (convert A to bf16) into LDS ----
#pragma unroll
    for (int i = 0; i < 8; ++i) {
      int fi = tid + 256 * i;
      int r  = fi >> 4;
      int c  = (fi & 15) * 4;
      unsigned short* dst = &As[r * LDA_S + c];
      dst[0] = f2bf(areg[i].x); dst[1] = f2bf(areg[i].y);
      dst[2] = f2bf(areg[i].z); dst[3] = f2bf(areg[i].w);
    }
#pragma unroll
    for (int i = 0; i < 2; ++i) {
      int fi = tid + 256 * i;
      int r  = fi >> 3;
      int c  = (fi & 7) * 8;
#pragma unroll
      for (int j = 0; j < 8; ++j) {
        unsigned short val =
            (unsigned short)((breg[i][j >> 1] >> ((j & 1) * 16)) & 0xFFFFu);
        Bs[(c + j) * LDB_S + r] = val;
      }
    }
    __syncthreads();

    // ---- issue global loads for next tile (overlap with WMMAs below) ----
    if (kt + BK < K) {
      const float*          Anext = Ag + (size_t)kt + BK;
      const unsigned short* Bnext = Bg + (size_t)(kt + BK) * ldb;
#pragma unroll
      for (int i = 0; i < 8; ++i) {
        int fi = tid + 256 * i;
        int r  = fi >> 4;
        int c  = (fi & 15) * 4;
        areg[i] = *(const float4*)(Anext + (size_t)(bm + r) * lda + c);
      }
#pragma unroll
      for (int i = 0; i < 2; ++i) {
        int fi = tid + 256 * i;
        int r  = fi >> 3;
        int c  = (fi & 7) * 8;
        breg[i] = *(const v4u*)(Bnext + (size_t)r * ldb + bn + c);
      }
      if (kt + 2 * BK < K)  // warm L2 for tile after next (global_prefetch_b8)
        __builtin_prefetch(Ag + (size_t)(bm + ar) * lda + kt + 2 * BK, 0, 1);
    }

    // ---- compute: 2 K-steps x (2x2) WMMA tiles per wave ----
#pragma unroll
    for (int ks = 0; ks < BK; ks += 32) {
      v16bf afrag[2], bfrag[2];
#pragma unroll
      for (int ti = 0; ti < 2; ++ti) {
        // A 16x32 frag: lane<16 holds row m, K {0..7,16..23}; lane>=16 K {8..15,24..31}
        const unsigned short* ap = &As[(wm + 16 * ti + l15) * LDA_S + ks];
        v4u lo = *(const v4u*)(ap + 8 * half);
        v4u hi = *(const v4u*)(ap + 16 + 8 * half);
        v8u u;
        u[0] = lo[0]; u[1] = lo[1]; u[2] = lo[2]; u[3] = lo[3];
        u[4] = hi[0]; u[5] = hi[1]; u[6] = hi[2]; u[7] = hi[3];
        afrag[ti] = __builtin_bit_cast(v16bf, u);
      }
#pragma unroll
      for (int tj = 0; tj < 2; ++tj) {
        // B 32x16 frag: lane<16 holds column n, K {0..7,16..23}; lane>=16 K {8..15,24..31}
        const unsigned short* bp = &Bs[(wn + 16 * tj + l15) * LDB_S + ks];
        v4u lo = *(const v4u*)(bp + 8 * half);
        v4u hi = *(const v4u*)(bp + 16 + 8 * half);
        v8u u;
        u[0] = lo[0]; u[1] = lo[1]; u[2] = lo[2]; u[3] = lo[3];
        u[4] = hi[0]; u[5] = hi[1]; u[6] = hi[2]; u[7] = hi[3];
        bfrag[tj] = __builtin_bit_cast(v16bf, u);
      }
#pragma unroll
      for (int ti = 0; ti < 2; ++ti)
#pragma unroll
        for (int tj = 0; tj < 2; ++tj)
          acc[ti][tj] = __builtin_amdgcn_wmma_f32_16x16x32_bf16(
              false, afrag[ti], false, bfrag[tj], (short)0, acc[ti][tj], false, false);
    }
  }

  // ---- store C: VGPR v, lanes 0-15 -> M=v, lanes 16-31 -> M=v+8 ----
#pragma unroll
  for (int ti = 0; ti < 2; ++ti)
#pragma unroll
    for (int tj = 0; tj < 2; ++tj) {
      int m0 = bm + wm + 16 * ti + 8 * half;
      int n0 = bn + wn + 16 * tj + l15;
#pragma unroll
      for (int v = 0; v < 8; ++v)
        Cg[(size_t)(m0 + v) * ldc + n0] = acc[ti][tj][v];
    }
}

// ---------------------------------------------------------------------------
// Persistent LSTM recurrence. 64 blocks x 8 waves; wave w of block b owns
// hidden index r = b*8+w (its 4 gate rows of W_hh and its W_out row).
// c kept in lane-0 registers, h exchanged via L2-resident hbuf + grid barrier.
// ---------------------------------------------------------------------------
__device__ __forceinline__ void grid_barrier(int* __restrict__ bar, int nblocks, int target) {
  __syncthreads();
  if (threadIdx.x == 0) {
    __threadfence();
    if (atomicAdd(&bar[0], 1) == nblocks - 1) {
      bar[0] = 0;
      __threadfence();
      atomicAdd(&bar[1], 1);
    } else {
      while (__hip_atomic_load(&bar[1], __ATOMIC_ACQUIRE, __HIP_MEMORY_SCOPE_AGENT) < target) {
        __builtin_amdgcn_s_sleep(1);
      }
    }
  }
  __syncthreads();
}

__global__ __launch_bounds__(256)
void lstm_kernel(const float* __restrict__ G, const float* __restrict__ Whh,
                 const float* __restrict__ bih, const float* __restrict__ bhh,
                 const float* __restrict__ Wout, const float* __restrict__ bout,
                 float* __restrict__ hbuf, int* __restrict__ bar,
                 float* __restrict__ out, int nblocks) {
  const int wave = threadIdx.x >> 5;
  const int lane = threadIdx.x & 31;
  const int r    = blockIdx.x * 8 + wave;   // 0..511
  float c = 0.0f;
  int gen = 0;

  for (int t = 0; t < T_STEPS; ++t) {
    // h_prev (16 floats / lane, as float4)
    float4 hv[4];
#pragma unroll
    for (int q = 0; q < 4; ++q) hv[q] = *(const float4*)&hbuf[lane * 4 + 128 * q];

    float gsum[4];
#pragma unroll
    for (int g = 0; g < 4; ++g) {          // 0=i 1=f 2=g 3=o
      const float* wrow = Whh + (size_t)(g * RDIM + r) * RDIM;
      float p = 0.0f;
#pragma unroll
      for (int q = 0; q < 4; ++q) {
        float4 w = *(const float4*)&wrow[lane * 4 + 128 * q];
        p += w.x * hv[q].x + w.y * hv[q].y + w.z * hv[q].z + w.w * hv[q].w;
      }
      gsum[g] = wave_reduce(p);
    }

    float hnew = 0.0f;
    if (lane == 0) {
      float gi = gsum[0] + G[(size_t)(0 * RDIM + r) * T_PAD + t] + bih[0 * RDIM + r] + bhh[0 * RDIM + r];
      float gf = gsum[1] + G[(size_t)(1 * RDIM + r) * T_PAD + t] + bih[1 * RDIM + r] + bhh[1 * RDIM + r];
      float gg = gsum[2] + G[(size_t)(2 * RDIM + r) * T_PAD + t] + bih[2 * RDIM + r] + bhh[2 * RDIM + r];
      float go = gsum[3] + G[(size_t)(3 * RDIM + r) * T_PAD + t] + bih[3 * RDIM + r] + bhh[3 * RDIM + r];
      c    = sigmoidf_(gf) * c + sigmoidf_(gi) * tanhf(gg);
      hnew = sigmoidf_(go) * tanhf(c);
    }

    grid_barrier(bar, nblocks, ++gen);      // all reads of h_prev done
    if (lane == 0) hbuf[r] = hnew;
    __threadfence();
    grid_barrier(bar, nblocks, ++gen);      // h_new visible everywhere

    // out[:, t] = W_out @ h_new + b_out
#pragma unroll
    for (int q = 0; q < 4; ++q) hv[q] = *(const float4*)&hbuf[lane * 4 + 128 * q];
    const float* worow = Wout + (size_t)r * RDIM;
    float p = 0.0f;
#pragma unroll
    for (int q = 0; q < 4; ++q) {
      float4 w = *(const float4*)&worow[lane * 4 + 128 * q];
      p += w.x * hv[q].x + w.y * hv[q].y + w.z * hv[q].z + w.w * hv[q].w;
    }
    p = wave_reduce(p);
    if (lane == 0) out[(size_t)r * T_STEPS + t] = p + bout[r];
  }
}

// ---------------------------------------------------------------------------
// Host launch
// ---------------------------------------------------------------------------
extern "C" void kernel_launch(void* const* d_in, const int* in_sizes, int n_in,
                              void* d_out, int out_size, void* d_ws, size_t ws_size,
                              hipStream_t stream) {
  const float* x    = (const float*)d_in[0];   // 512 x 500
  const float* A    = (const float*)d_in[1];   // 512 x 512
  const float* Wgc  = (const float*)d_in[2];   // 32 x 1
  const float* bgc  = (const float*)d_in[3];   // 32
  const float* Wih  = (const float*)d_in[4];   // 2048 x 16384
  const float* bih  = (const float*)d_in[5];   // 2048
  const float* Whh  = (const float*)d_in[6];   // 2048 x 512
  const float* bhh  = (const float*)d_in[7];   // 2048
  const float* Wout = (const float*)d_in[8];   // 512 x 512
  const float* bout = (const float*)d_in[9];   // 512
  float* out = (float*)d_out;                  // 512 x 500

  // workspace layout (~22.6 MB, all offsets 256B aligned)
  char* ws = (char*)d_ws;
  unsigned short* Xbf = (unsigned short*)(ws + 0);          //  512*512*2  = 0.5 MB
  unsigned short* Inp = (unsigned short*)(ws + 524288);     // 16384*512*2 = 16 MB
  float* Y    = (float*)(ws + 17301504);                    //  512*512*4  = 1 MB
  float* G    = (float*)(ws + 18350080);                    // 2048*512*4  = 4 MB
  float* s    = (float*)(ws + 22544384);                    //  512*4
  float* hbuf = (float*)(ws + 22546432);                    //  512*4
  int*   bar  = (int*)  (ws + 22548480);                    //  2 ints

  init_kernel<<<1, 512, 0, stream>>>(hbuf, bar);
  convert_x_kernel<<<(512 * 512) / 256, 256, 0, stream>>>(x, Xbf);
  rowsum_kernel<<<512, 256, 0, stream>>>(A, s);
  // Y = A @ Xpad   (M=512, N=512, K=512)
  gemm_bf16_kernel<<<dim3(512 / BM, 512 / BN), 256, 0, stream>>>(A, Xbf, Y, 512, 512, 512, 512);
  expand_kernel<<<dim3(T_PAD / 128, N_NODES), 128, 0, stream>>>(Y, s, Wgc, bgc, Inp);
  // G = W_ih @ Inp (M=2048, N=512, K=16384)  -- dominant GEMM
  gemm_bf16_kernel<<<dim3(2048 / BM, 512 / BN), 256, 0, stream>>>(Wih, Inp, G, KBIG, KBIG, T_PAD, T_PAD);
  lstm_kernel<<<64, 256, 0, stream>>>(G, Whh, bih, bhh, Wout, bout, hbuf, bar, out, 64);
}
